// SDPA_74062416052399
// MI455X (gfx1250) — matrix operand order repaired and checked by reading, not compiled, not run on hardware
//
#include <hip/hip_runtime.h>

#define DEV static __device__ __forceinline__

typedef __attribute__((ext_vector_type(16))) __bf16 v16bf;
typedef __attribute__((ext_vector_type(8)))  __bf16 bf16x8;
typedef __attribute__((ext_vector_type(4)))  __bf16 bf16x4;
typedef __attribute__((ext_vector_type(8)))  float  v8f;
typedef __attribute__((ext_vector_type(8)))  unsigned int u32x8;
typedef __attribute__((ext_vector_type(4)))  unsigned int u32x4v;
typedef __attribute__((ext_vector_type(4)))  int i32x4v;
typedef __attribute__((ext_vector_type(8)))  int i32x8v;

constexpr int BM  = 32;    // B*M
constexpr int QT  = 2048;
constexpr int KVT = 2048;
constexpr int H   = 128;

constexpr unsigned KBUF_BYTES = 32 * H * 2;   // 8 KB : 32 kv rows x 128 h (bf16)
constexpr unsigned VBUF_BYTES = H * 32 * 2;   // 8 KB : 128 h rows x 32 kv (bf16)
constexpr unsigned LDS_K0 = 0;
constexpr unsigned LDS_V0 = 2 * KBUF_BYTES;   // 16384

// ---------------- helpers ----------------

DEV bf16x8 ld8(const __bf16* p) { return *(const bf16x8*)p; }

DEV v16bf combine(bf16x8 lo, bf16x8 hi) {
  v16bf r;
#pragma unroll
  for (int i = 0; i < 8; ++i) { r[i] = lo[i]; r[8 + i] = hi[i]; }
  return r;
}

DEV v16bf cvt16(const float* p) {
  v16bf r;
#pragma unroll
  for (int j = 0; j < 4; ++j) {
    float4 f = *(const float4*)(p + j * 4);
    r[j*4+0] = (__bf16)f.x; r[j*4+1] = (__bf16)f.y;
    r[j*4+2] = (__bf16)f.z; r[j*4+3] = (__bf16)f.w;
  }
  return r;
}

DEV unsigned pack_bf16(float a, float b) {
  __bf16 x = (__bf16)a, y = (__bf16)b;
  unsigned short xu = __builtin_bit_cast(unsigned short, x);
  unsigned short yu = __builtin_bit_cast(unsigned short, y);
  return (unsigned)xu | ((unsigned)yu << 16);
}

DEV v8f wmma_bf16(v16bf a, v16bf b, v8f c) {
  return __builtin_amdgcn_wmma_f32_16x16x32_bf16(
      false, a, false, b, (short)0, c, false, false);
}

// ---- Tensor Data Mover: 2-D tile (data_size = 2B) global -> LDS ----
// D# per CDNA5 ISA. Groups 2/3 zero (<=2-D tensor).
DEV void tdm_load_2d(unsigned lds_off, unsigned long gaddr,
                     unsigned tile_d0, unsigned tile_d1,
                     unsigned tensor_d0, unsigned tensor_d1,
                     unsigned long stride0_elems) {
  u32x4v g0;
  g0[0] = 1u;                                        // count=1, user descriptor
  g0[1] = lds_off;                                   // lds_addr (bytes)
  g0[2] = (unsigned)(gaddr & 0xFFFFFFFFu);           // global_addr[31:0]
  g0[3] = (unsigned)((gaddr >> 32) & 0x1FFFFFFu)     // global_addr[56:32]
          | (2u << 30);                              // type = 2 (image)
  i32x8v g1;
  g1[0] = (int)(1u << 16);                           // data_size=1 -> 2 bytes
  g1[1] = (int)((tensor_d0 & 0xFFFFu) << 16);        // tensor_dim0[15:0] @bit48
  g1[2] = (int)((tensor_d0 >> 16) | ((tensor_d1 & 0xFFFFu) << 16));
  g1[3] = (int)((tensor_d1 >> 16) | (tile_d0 << 16));// tile_dim0 @bit112
  g1[4] = (int)(tile_d1 & 0xFFFFu);                  // tile_dim1; tile_dim2=0
  g1[5] = (int)(unsigned)(stride0_elems & 0xFFFFFFFFu); // dim0 stride[31:0]
  g1[6] = (int)(unsigned)((stride0_elems >> 32) & 0xFFFFu);
  g1[7] = 0;
  i32x4v z4 = {0, 0, 0, 0};
#if __clang_major__ >= 23
  i32x8v z8 = {0, 0, 0, 0, 0, 0, 0, 0};
  __builtin_amdgcn_tensor_load_to_lds(g0, g1, z4, z4, z8, 0);
#else
  __builtin_amdgcn_tensor_load_to_lds(g0, g1, z4, z4, 0);
#endif
}

// ---------------- prologue: K fp32 -> bf16 (same layout) ----------------

__global__ __launch_bounds__(256) void cvt_k_kernel(const float* __restrict__ k,
                                                    __bf16* __restrict__ kb, int n) {
  int i = (blockIdx.x * 256 + threadIdx.x) * 4;
  if (i < n) {
    float4 f = *(const float4*)(k + i);
    bf16x4 o = { (__bf16)f.x, (__bf16)f.y, (__bf16)f.z, (__bf16)f.w };
    *(bf16x4*)(kb + i) = o;
  }
}

// ---------------- prologue: V fp32 [bm][kv][h] -> bf16 V^T [bm][h][kv] ----

__global__ __launch_bounds__(256) void tr_v_kernel(const float* __restrict__ v,
                                                   __bf16* __restrict__ vt) {
  __shared__ __bf16 tile[32][33];
  const int bm = blockIdx.z;
  const int kv0 = blockIdx.x * 32, h0 = blockIdx.y * 32;
  const int tx = threadIdx.x, ty = threadIdx.y;
  const float* src = v + ((size_t)bm * KVT + kv0) * H + h0;
#pragma unroll
  for (int j = 0; j < 4; ++j) {
    int kv = ty + j * 8;
    tile[kv][tx] = (__bf16)src[(size_t)kv * H + tx];
  }
  __syncthreads();
  __bf16* dst = vt + ((size_t)bm * H + h0) * KVT + kv0;
#pragma unroll
  for (int j = 0; j < 4; ++j) {
    int h = ty + j * 8;
    dst[(size_t)h * KVT + tx] = tile[tx][h];
  }
}

// ---------------- main: transposed flash attention + TDM staging ---------
// grid: (QT/128, BM), block 256 = 8 waves; wave w owns q rows [qb, qb+16)
// S^T = K·Q^T (M=kv,N=q,K=h) ; O^T = V^T·P^T (M=h,N=q,K=kv)
// K/V^T chunks DMA'd to LDS by wave 0 via TENSOR_LOAD_TO_LDS, double-buffered.

__global__ __launch_bounds__(256, 1) void fattn_kernel(
    const float* __restrict__ q, const __bf16* __restrict__ kb,
    const __bf16* __restrict__ vt, float* __restrict__ out) {
  __shared__ __align__(1024) char smem[2 * KBUF_BYTES + 2 * VBUF_BYTES]; // 32 KB

  const int bm   = blockIdx.y;
  const int wave = threadIdx.x >> 5;
  const int lane = threadIdx.x & 31;
  const int l16  = lane & 15;
  const int half = lane >> 4;
  const int q_base = blockIdx.x * 128 + wave * 16;

  // Q^T B-fragments: lane holds Q[q_base+l16][c*32 + half*16 .. +16)
  const float* qrow = q + ((size_t)bm * QT + q_base + l16) * (size_t)H;
  v16bf qf[4];
#pragma unroll
  for (int c = 0; c < 4; ++c) qf[c] = cvt16(qrow + c * 32 + half * 16);

  v8f acc[8];
#pragma unroll
  for (int t = 0; t < 8; ++t) acc[t] = (v8f){0,0,0,0,0,0,0,0};
  float m_run = -3.0e38f, l_run = 0.0f;

  const unsigned long kg = (unsigned long)(const void*)(kb + (size_t)bm * KVT * H);
  const unsigned long vg = (unsigned long)(const void*)(vt + (size_t)bm * H * KVT);
  constexpr unsigned BIG = 1u << 30;
  constexpr int NCH = KVT / 32;  // 64 chunks

  if (wave == 0) {
    // chunk 0: K tile is one contiguous 8KB run; V^T tile is 128 rows x 64B.
    tdm_load_2d(LDS_K0, kg, 4096, 1, BIG, 1, 4096);
    tdm_load_2d(LDS_V0, vg, 32, 128, BIG, BIG, (unsigned long)KVT);
    __builtin_amdgcn_s_wait_tensorcnt(0);
  }
  __syncthreads();

  for (int it = 0; it < NCH; ++it) {
    const int buf = it & 1;
    if (wave == 0 && it + 1 < NCH) {
      const unsigned long kvn = (unsigned long)(it + 1) * 32;
      tdm_load_2d(LDS_K0 + (1 - buf) * KBUF_BYTES, kg + kvn * H * 2,
                  4096, 1, BIG, 1, 4096);
      tdm_load_2d(LDS_V0 + (1 - buf) * VBUF_BYTES, vg + kvn * 2,
                  32, 128, BIG, BIG, (unsigned long)KVT);
    }

    const char* kbuf = smem + LDS_K0 + buf * KBUF_BYTES; // [kv 32][h 128] bf16
    const char* vbuf = smem + LDS_V0 + buf * VBUF_BYTES; // [h 128][kv 32] bf16

    // ---- S^T tiles: s0 = kv[0..16), s1 = kv[16..32) ----
    v8f s0 = (v8f){0,0,0,0,0,0,0,0};
    v8f s1 = (v8f){0,0,0,0,0,0,0,0};
    const char* krow0 = kbuf + l16 * 256;        // kv = l16
    const char* krow1 = kbuf + (16 + l16) * 256; // kv = 16 + l16
#pragma unroll
    for (int c = 0; c < 4; ++c) {
      v16bf kf0 = combine(ld8((const __bf16*)(krow0 + c * 64 + half * 32)),
                          ld8((const __bf16*)(krow0 + c * 64 + half * 32 + 16)));
      s0 = wmma_bf16(kf0, qf[c], s0);
      v16bf kf1 = combine(ld8((const __bf16*)(krow1 + c * 64 + half * 32)),
                          ld8((const __bf16*)(krow1 + c * 64 + half * 32 + 16)));
      s1 = wmma_bf16(kf1, qf[c], s1);
    }

    // ---- online softmax: row (q) index == lane index ----
    float mloc = s0[0];
#pragma unroll
    for (int r = 0; r < 8; ++r) { mloc = fmaxf(mloc, s0[r]); mloc = fmaxf(mloc, s1[r]); }
    mloc = fmaxf(mloc, __shfl_xor(mloc, 16, 32));
    float mnew = fmaxf(m_run, mloc);
    float corr = __expf(m_run - mnew);

    float e0[8], e1[8], ssum = 0.0f;
#pragma unroll
    for (int r = 0; r < 8; ++r) {
      e0[r] = __expf(s0[r] - mnew);
      e1[r] = __expf(s1[r] - mnew);
      ssum += e0[r] + e1[r];
    }
    ssum += __shfl_xor(ssum, 16, 32);
    l_run = l_run * corr + ssum;
    m_run = mnew;

#pragma unroll
    for (int t = 0; t < 8; ++t)
#pragma unroll
      for (int j = 0; j < 8; ++j) acc[t][j] *= corr;

    // ---- build P^T B-fragment (kv x q) via half-wave swaps ----
    unsigned p0[4], p1[4];
#pragma unroll
    for (int j = 0; j < 4; ++j) {
      p0[j] = pack_bf16(e0[2 * j], e0[2 * j + 1]);
      p1[j] = pack_bf16(e1[2 * j], e1[2 * j + 1]);
    }
    u32x8 d;
#pragma unroll
    for (int vv = 0; vv < 4; ++vv) {
      unsigned o1 = __shfl_xor(p1[vv], 16, 32);
      unsigned o0 = __shfl_xor(p0[vv], 16, 32);
      d[vv]     = half ? o1 : p0[vv];  // kv 0-7 (half 0), 16-23 (half 1)
      d[vv + 4] = half ? p1[vv] : o0;  // kv 8-15 (half 0), 24-31 (half 1)
    }
    v16bf pf = __builtin_bit_cast(v16bf, d);

    // ---- O^T += V^T · P^T over 8 h-tiles ----
#pragma unroll
    for (int t = 0; t < 8; ++t) {
      const char* vrow = vbuf + (t * 16 + l16) * 64;
      v16bf vf = combine(ld8((const __bf16*)(vrow + half * 32)),
                         ld8((const __bf16*)(vrow + half * 32 + 16)));
      acc[t] = wmma_bf16(vf, pf, acc[t]);
    }

    if (wave == 0) __builtin_amdgcn_s_wait_tensorcnt(0);
    __syncthreads();
  }

  // ---- epilogue: normalize + store (contiguous h runs per lane) ----
  const float inv = 1.0f / l_run;
  float* orow = out + ((size_t)bm * QT + q_base + l16) * (size_t)H;
#pragma unroll
  for (int t = 0; t < 8; ++t) {
    int h0 = t * 16 + half * 8;
    float4 lo = { acc[t][0] * inv, acc[t][1] * inv, acc[t][2] * inv, acc[t][3] * inv };
    float4 hi = { acc[t][4] * inv, acc[t][5] * inv, acc[t][6] * inv, acc[t][7] * inv };
    *(float4*)(orow + h0)     = lo;
    *(float4*)(orow + h0 + 4) = hi;
  }
}

// ---------------- launch ----------------

extern "C" void kernel_launch(void* const* d_in, const int* in_sizes, int n_in,
                              void* d_out, int out_size, void* d_ws, size_t ws_size,
                              hipStream_t stream) {
  const float* q = (const float*)d_in[0];
  const float* k = (const float*)d_in[1];
  const float* v = (const float*)d_in[2];
  float* out = (float*)d_out;

  const size_t nkv = (size_t)BM * KVT * H;  // 8,388,608 elements
  __bf16* kb = (__bf16*)d_ws;
  __bf16* vt = (__bf16*)((char*)d_ws + nkv * sizeof(__bf16));

  cvt_k_kernel<<<(int)(nkv / (256 * 4)), 256, 0, stream>>>(k, kb, (int)nkv);

  dim3 tg(KVT / 32, H / 32, BM);
  tr_v_kernel<<<tg, dim3(32, 8), 0, stream>>>(v, vt);

  fattn_kernel<<<dim3(QT / 128, BM), 256, 0, stream>>>(q, kb, vt, out);
}
